// GNNModel_63204738728140
// MI455X (gfx1250) — matrix operand order repaired and checked by reading, not compiled, not run on hardware
//
#include <hip/hip_runtime.h>

// ---------------- WMMA types (CDNA5 gfx1250, wave32) ----------------
typedef __attribute__((ext_vector_type(16))) __bf16 v16bf_t;
typedef __attribute__((ext_vector_type(8)))  float  v8f_t;

union FragAB { unsigned u[8]; v16bf_t v; };
union FragC  { float    f[8]; v8f_t   v; };

// LDS pitches (elements). Chosen so fragment loads are 16B-aligned and row
// strides are NOT multiples of 64 dwords (avoids full-bank-conflict serialization).
#define AP 136   // bf16 A buffer pitch (272 B/row)
#define WP 136   // bf16 W buffer pitch (272 B/row)
#define FP 132   // f32 Msg/Relq pitch  (528 B/row)

// Pack two f32 -> packed bf16 pair with round-to-nearest via +0x8000, then a
// single v_perm_b32 grabs the two high halves. 3 VALU ops per pair.
__device__ __forceinline__ unsigned pack_bf16(float a, float b) {
    unsigned ua = __float_as_uint(a) + 0x8000u;
    unsigned ub = __float_as_uint(b) + 0x8000u;
    return __builtin_amdgcn_perm(ub, ua, 0x07060302u);
}

// Intra-wave LDS fence: DS ops from one wave complete in order; this waits
// for outstanding DS stores and stops compiler reordering.
__device__ __forceinline__ void lds_fence() {
    asm volatile("s_wait_dscnt 0" ::: "memory");
}

// A fragment (16x32 bf16, K-chunk c) from a row-major packed-bf16 LDS buffer
// (pitch AP). ISA layout: lane m holds row m K={c*32+0..7, +16..23}; lane m+16
// holds row m K={c*32+8..15, +24..31}. Pure ds_load_b128, no VALU.
__device__ __forceinline__ void load_afrag(const unsigned short* ab, int lane, int c, FragAB& f) {
    const int m = lane & 15, half = lane >> 4;
    const unsigned short* base = ab + m * AP + c * 32 + half * 8;
    uint4 a = *(const uint4*)(base);
    uint4 b = *(const uint4*)(base + 16);
    f.u[0] = a.x; f.u[1] = a.y; f.u[2] = a.z; f.u[3] = a.w;
    f.u[4] = b.x; f.u[5] = b.y; f.u[6] = b.z; f.u[7] = b.w;
}

// B fragment (32x16 bf16) for K-chunk c, N-slab j from row-major bf16 W copy.
// Lane n holds column N=j*16+n%16, K={c*32+8h+0..7, +16..23} with h=n/16.
__device__ __forceinline__ void load_bfrag(const unsigned short* wl, int lane, int c, int j, FragAB& bf) {
    const unsigned short* base = wl + (c * 32 + (lane >> 4) * 8) * WP + j * 16 + (lane & 15);
#pragma unroll
    for (int i = 0; i < 8; ++i) {
        int k = (i < 4) ? 2 * i : 16 + 2 * (i - 4);
        bf.u[i] = (unsigned)base[k * WP] | ((unsigned)base[(k + 1) * WP] << 16);
    }
}

// Stage a 128x128 f32 W from global into LDS as bf16 with padded pitch WP.
// 256 threads x 32 pairs.
__device__ __forceinline__ void stage_w(const float* __restrict__ W, unsigned short* wl, int t) {
    const float2* wf = (const float2*)W;
    unsigned* wp = (unsigned*)wl;
#pragma unroll
    for (int k = 0; k < 32; ++k) {
        int p = t * 32 + k;            // pair id 0..8191
        int row = p >> 6, col2 = p & 63;
        float2 v = wf[row * 64 + col2];
        wp[row * (WP / 2) + col2] = pack_bf16(v.x, v.y);
    }
}

// ---------------- small helper kernels ----------------
__global__ void zero_kernel(float4* p, long long n4) {
    long long i = (long long)blockIdx.x * blockDim.x + threadIdx.x;
    if (i < n4) p[i] = make_float4(0.f, 0.f, 0.f, 0.f);
}

__global__ void gather_qsel_kernel(const float* rel, const int* qr, float* qsel, int B) {
    int i = blockIdx.x * blockDim.x + threadIdx.x;
    if (i < B * 128) {
        int row = i >> 7, col = i & 127;
        qsel[i] = rel[(size_t)(qr[row] + 1) * 128 + col];
    }
}

// ---------------- generic (M,128) @ (128,128) + bias, bf16 WMMA ----------------
__global__ __launch_bounds__(256) void gemm128_kernel(const float* __restrict__ A, int M,
                                                      const float* __restrict__ W,
                                                      const float* __restrict__ bias,
                                                      float* __restrict__ Out) {
    __shared__ unsigned short As[16 * AP];
    __shared__ unsigned short Wl[128 * WP];
    __shared__ float          biasl[128];
    const int t = threadIdx.x;

    stage_w(W, Wl, t);
    if (t < 128) biasl[t] = bias ? bias[t] : 0.f;
    {   // stage A tile as packed bf16 (zero-padded past M)
        int row = t >> 4, sub = t & 15;
        int g = blockIdx.x * 16 + row;
        float4 z = make_float4(0.f, 0.f, 0.f, 0.f), v0 = z, v1 = z;
        if (g < M) { const float4* ap = (const float4*)(A + (size_t)g * 128 + sub * 8); v0 = ap[0]; v1 = ap[1]; }
        uint4 pk;
        pk.x = pack_bf16(v0.x, v0.y); pk.y = pack_bf16(v0.z, v0.w);
        pk.z = pack_bf16(v1.x, v1.y); pk.w = pack_bf16(v1.z, v1.w);
        *(uint4*)(As + row * AP + sub * 8) = pk;
    }
    __syncthreads();

    const int lane = t & 31, j = t >> 5;         // 8 waves -> 8 N-slabs
    const int m = lane & 15, half = lane >> 4;
    FragC c;
    float bv = biasl[j * 16 + m];
#pragma unroll
    for (int r = 0; r < 8; ++r) c.f[r] = bv;
#pragma unroll
    for (int cc = 0; cc < 4; ++cc) {
        FragAB afr, bfr;
        load_afrag(As, lane, cc, afr);
        load_bfrag(Wl, lane, cc, j, bfr);
        c.v = __builtin_amdgcn_wmma_f32_16x16x32_bf16(false, afr.v, false, bfr.v, (short)0, c.v, false, false);
    }
    const int g0 = blockIdx.x * 16;
    float* op = Out + (size_t)(g0 + half * 8) * 128 + j * 16 + m;
    if (g0 + 16 <= M) {                          // fast path: full tile, branchless
#pragma unroll
        for (int r = 0; r < 8; ++r) op[(size_t)r * 128] = c.f[r];
    } else {
#pragma unroll
        for (int r = 0; r < 8; ++r) {
            int g = g0 + r + half * 8;
            if (g < M) Out[(size_t)g * 128 + j * 16 + m] = c.f[r];
        }
    }
}

// ---------------- fused per-fact kernel ----------------
// pre = h_e@Ws + (relproj[r+1] + qproj[b]);  alpha = sigmoid(relu(pre)@wa + wab)
// atomic scatter of alpha*(h_e+hr) into agg[tail].
#define ABUF_B (16 * AP * 2)                    // 4352: packed-bf16 h_e rows
#define MSG_B  (16 * FP * 4)                    // 8448: f32 h_e+hr
#define RELQ_B (16 * FP * 4)                    // 8448: f32 relproj+qproj
#define WV_BYTES   (ABUF_B + MSG_B + RELQ_B + 64)
#define SMEM_BYTES (512 + 128 * WP * 2 + 8 * WV_BYTES)

__global__ __launch_bounds__(256) void fact_kernel(const float* __restrict__ hidden,
                                                   const float* __restrict__ rel,
                                                   const float* __restrict__ Ws,
                                                   const float* __restrict__ wa_w,
                                                   const float* __restrict__ wa_b,
                                                   const int*   __restrict__ facts,
                                                   const int*   __restrict__ tail,
                                                   const float* __restrict__ relproj,
                                                   const float* __restrict__ qproj,
                                                   float* __restrict__ agg,
                                                   long long E, long long numTiles) {
    extern __shared__ char smem[];
    float*          wal = (float*)smem;
    unsigned short* wsl = (unsigned short*)(smem + 512);
    const int t = threadIdx.x, waveId = t >> 5, lane = t & 31;
    char*  wb     = smem + 512 + 128 * WP * 2 + waveId * WV_BYTES;
    unsigned short* Abuf = (unsigned short*)wb;            // packed bf16 h_e (16 x AP)
    float* Msg  = (float*)(wb + ABUF_B);                   // h_e + hr        (16 x FP)
    float* Relq = (float*)(wb + ABUF_B + MSG_B);           // relproj + qproj (16 x FP)
    float* ab   = (float*)(wb + ABUF_B + MSG_B + RELQ_B);  // alpha[16]

    stage_w(Ws, wsl, t);
    if (t < 128) wal[t] = wa_w[t];
    __syncthreads();
    const float wab = wa_b[0];

    const int m16 = lane & 15, half16 = lane >> 4;  // WMMA fragment coords
    const int row2 = lane >> 1, half2 = lane & 1;   // 2 lanes per fact row for staging

    long long gw = (long long)blockIdx.x * 8 + waveId;
    long long nw = (long long)gridDim.x * 8;
    for (long long tile = gw; tile < numTiles; tile += nw) {
        long long f0 = tile * 16;
        int nI = 0, rI = 0, bI = 0, tI = 0, vI = 0;
        if (lane < 16) {
            long long f = f0 + lane;
            if (f < E) {
                const int* fp = facts + f * 6;
                bI = fp[0]; nI = fp[1]; rI = fp[3]; tI = tail[f]; vI = 1;
            }
            // ---- prefetch next grid-stride tile's hidden rows ----
            // The facts[] index load below also warms next iteration's fact
            // cachelines; global_prefetch_b8 pulls the 4 lines of the 512B
            // hidden row so the next gather hits WGP$/L2 (only ~2 waves/SIMD
            // at this LDS footprint, so SW prefetch supplies the latency
            // hiding occupancy can't).
            long long fN = f0 + nw * 16 + lane;
            if (fN < E) {
                int nN = facts[fN * 6 + 1];
                const char* hp = (const char*)(hidden + (size_t)nN * 128);
                __builtin_prefetch(hp, 0, 1);
                __builtin_prefetch(hp + 128, 0, 1);
                __builtin_prefetch(hp + 256, 0, 1);
                __builtin_prefetch(hp + 384, 0, 1);
            }
        }
        {   // cooperative gather: each lane covers half a row (64 floats)
            int n = __shfl(nI, row2), r = __shfl(rI, row2), b = __shfl(bI, row2);
            const float4* hp = (const float4*)(hidden + (size_t)n * 128) + half2 * 16;
            const float4* rp = (const float4*)(rel + (size_t)(r + 1) * 128) + half2 * 16;
            const float4* pp = (const float4*)(relproj + (size_t)(r + 1) * 128) + half2 * 16;
            const float4* qp = (const float4*)(qproj + (size_t)b * 128) + half2 * 16;
            uint2*  Ad = (uint2*)(Abuf + row2 * AP + half2 * 64);
            float4* Md = (float4*)(Msg + row2 * FP) + half2 * 16;
            float4* Rd = (float4*)(Relq + row2 * FP) + half2 * 16;
#pragma unroll
            for (int i = 0; i < 16; ++i) {
                float4 h = hp[i], hr = rp[i], pr = pp[i], q = qp[i];
                uint2 pk;
                pk.x = pack_bf16(h.x, h.y); pk.y = pack_bf16(h.z, h.w);
                Ad[i] = pk;
                Md[i] = make_float4(h.x + hr.x, h.y + hr.y, h.z + hr.z, h.w + hr.w);
                Rd[i] = make_float4(pr.x + q.x, pr.y + q.y, pr.z + q.z, pr.w + q.w);
            }
        }
        lds_fence();

        float p8[8];
#pragma unroll
        for (int r = 0; r < 8; ++r) p8[r] = 0.f;

#pragma unroll
        for (int j = 0; j < 8; ++j) {               // 8 N-slabs of 16 cols
            FragC c;
#pragma unroll
            for (int r = 0; r < 8; ++r)             // C init = relproj+qproj term
                c.f[r] = Relq[(r + half16 * 8) * FP + j * 16 + m16];
#pragma unroll
            for (int cc = 0; cc < 4; ++cc) {
                FragAB afr, bfr;
                load_afrag(Abuf, lane, cc, afr);
                load_bfrag(wsl, lane, cc, j, bfr);
                c.v = __builtin_amdgcn_wmma_f32_16x16x32_bf16(false, afr.v, false, bfr.v,
                                                              (short)0, c.v, false, false);
            }
            float w = wal[j * 16 + m16];            // fold wa dot into slab pass
#pragma unroll
            for (int r = 0; r < 8; ++r) p8[r] += fmaxf(c.f[r], 0.f) * w;
        }
        // butterfly within 16-lane halves -> alpha per fact row
#pragma unroll
        for (int r = 0; r < 8; ++r) {
            float s = p8[r];
            s += __shfl_xor(s, 1); s += __shfl_xor(s, 2);
            s += __shfl_xor(s, 4); s += __shfl_xor(s, 8);
            p8[r] = 1.f / (1.f + __expf(-(s + wab)));
        }
        if (m16 == 0) {                              // lanes 0 and 16 publish alpha
#pragma unroll
            for (int r = 0; r < 8; ++r) ab[half16 * 8 + r] = p8[r];
        }
        lds_fence();

        {   // scatter message with f32 atomics (agg is L2-resident: 51 MB < 192 MB)
            int valid = __shfl(vI, row2);
            if (valid) {
                int tt = __shfl(tI, row2);
                float al = ab[row2];
                float* dst = agg + (size_t)tt * 128 + half2 * 64;
                const float4* mp = (const float4*)(Msg + row2 * FP) + half2 * 16;
#pragma unroll
                for (int i = 0; i < 16; ++i) {
                    float4 v = mp[i];
                    atomicAdd(dst + i * 4 + 0, al * v.x);
                    atomicAdd(dst + i * 4 + 1, al * v.y);
                    atomicAdd(dst + i * 4 + 2, al * v.z);
                    atomicAdd(dst + i * 4 + 3, al * v.w);
                }
            }
        }
    }
}

// ---------------- host launcher ----------------
extern "C" void kernel_launch(void* const* d_in, const int* in_sizes, int n_in,
                              void* d_out, int out_size, void* d_ws, size_t ws_size,
                              hipStream_t stream) {
    const float* hidden    = (const float*)d_in[0];
    const float* rel_table = (const float*)d_in[1];
    const float* Ws        = (const float*)d_in[2];
    const float* Wr        = (const float*)d_in[3];
    const float* Wqr_w     = (const float*)d_in[4];
    const float* Wqr_b     = (const float*)d_in[5];
    const float* wa_w      = (const float*)d_in[6];
    const float* wa_b      = (const float*)d_in[7];
    const float* Wh        = (const float*)d_in[8];
    const int*   query_rel = (const int*)d_in[9];
    const int*   facts     = (const int*)d_in[10];
    const int*   tail_idx  = (const int*)d_in[11];

    const int NR = in_sizes[1] / 128;      // 2R+1 = 481
    const int Bq = in_sizes[9];            // 512
    const long long E = in_sizes[10] / 6;  // 1,000,000
    const int T  = in_sizes[12];           // 100,000

    float* ws      = (float*)d_ws;
    float* agg     = ws;
    float* relproj = agg + (size_t)T * 128;
    float* qproj   = relproj + (size_t)NR * 128;
    float* qsel    = qproj + (size_t)Bq * 128;
    float* out     = (float*)d_out;

    // 1) zero agg
    long long n4 = (long long)T * 32;
    zero_kernel<<<(unsigned)((n4 + 255) / 256), 256, 0, stream>>>((float4*)agg, n4);

    // 2) gather rel_table rows for query_rel
    gather_qsel_kernel<<<(Bq * 128 + 255) / 256, 256, 0, stream>>>(rel_table, query_rel, qsel, Bq);

    // 3) relproj = rel_table @ Wr ; qproj = qsel @ Wqr_w + Wqr_b
    gemm128_kernel<<<(NR + 15) / 16, 256, 0, stream>>>(rel_table, NR, Wr, nullptr, relproj);
    gemm128_kernel<<<(Bq + 15) / 16, 256, 0, stream>>>(qsel, Bq, Wqr_w, Wqr_b, qproj);

    // 4) fused gather + WMMA + alpha + atomic scatter over all facts
    long long numTiles = (E + 15) / 16;
    hipFuncSetAttribute((const void*)fact_kernel,
                        hipFuncAttributeMaxDynamicSharedMemorySize, SMEM_BYTES);
    fact_kernel<<<2048, 256, SMEM_BYTES, stream>>>(hidden, rel_table, Ws, wa_w, wa_b,
                                                   facts, tail_idx, relproj, qproj, agg,
                                                   E, numTiles);

    // 5) out = agg @ Wh
    gemm128_kernel<<<(T + 15) / 16, 256, 0, stream>>>(agg, T, Wh, nullptr, out);
}
